// HGTWithMLP_35742717837579
// MI455X (gfx1250) — matrix operand order repaired and checked by reading, not compiled, not run on hardware
//
#include <hip/hip_runtime.h>
#include <hip/hip_bf16.h>

// ---------------------------------------------------------------------------
// HGT (2-type heterogeneous graph transformer) + link MLP for MI455X (gfx1250)
// Dense GEMMs -> v_wmma_f32_16x16x32_f16 (wave32 WMMA, f32 accumulate)
//   - LDS tiles stored in fragment order: each lane loads its v16h operand
//     as 32 contiguous bytes (2x ds_load_b128).
//   - staging uses clamped addresses (no divergent bounds branches): OOB A
//     rows / B cols only affect C rows/cols that are never stored.
//   - A tile staged as float2 pairs -> packed 2xf16 ds_store_b32.
//   - gather (concat-embedding) variant split out via template.
// Edge softmax/scatter -> VALU + global float atomics
// Requires: K % 32 == 0 for every GEMM call (true for all call sites).
// ---------------------------------------------------------------------------

typedef _Float16 h16;
typedef __attribute__((ext_vector_type(16))) _Float16 v16h;
typedef __attribute__((ext_vector_type(8)))  float    v8f;

#define TILE_M 64
#define TILE_N 32
#define TILE_K 32
#define HID    64
#define NHEAD  2
#define DHEAD  32

// epilogue modes
#define EP_BIAS 0
#define EP_RELU 1
#define EP_SKIP 2   // D = sigmoid(gate)*(acc+bias) + (1-sigmoid(gate))*old

__device__ __forceinline__ int iclamp(int v, int hi) { return v < hi ? v : hi; }

// ---------------------------------------------------------------------------
// Tiled WMMA GEMM:  C[M,N] = act( A[M,K] @ B[K,N] + bias )
// GATHER: A row r comes from A[idx0[r]] (col < splitK) or A1[idx1[r]] (else).
// Block: 256 threads = 8 waves, 4x2 wave grid -> 64x32 output tile.
//
// LDS layout is fragment-ready:
//   As[rb][lane][e], rb = wave-row (16-row block), e = v16h element index.
//   A 16x32 fp16 ISA layout: lane = m%16 + 16*hl, element e holds
//     k = (e<8?0:16) + 8*hl + (e&7)   =>  inverse: hl=(k>>3)&1,
//     lane = (m&15)+16*hl, e = (k&7) + (k&16 ? 8 : 0)   (k,k+1 adjacent for even k)
//   B 32x16 fp16 ISA layout: lane = n + 16*hl, element e holds k = (e<8?0:16) + 8*hl + (e&7)
//     =>  inverse: lane = (n&15) + 16*((k>>3)&1), e = (k&7) + (k&16 ? 8 : 0)
// ---------------------------------------------------------------------------
template<bool GATHER>
__global__ __launch_bounds__(256)
void gemm_wmma_kernel(const float* __restrict__ A, int lda,
                      const float* __restrict__ A1,
                      const int* __restrict__ idx0, const int* __restrict__ idx1,
                      int splitK,
                      const float* __restrict__ B, int ldb,
                      const float* __restrict__ bias,
                      float* __restrict__ C, int ldc,
                      int M, int N, int K,
                      int epilogue,
                      const float* __restrict__ oldx, int ldold,
                      const float* __restrict__ gate)
{
    __shared__ alignas(32) h16 As[4][32][16];   // 4 KB, fragment order
    __shared__ alignas(32) h16 Bs[2][32][16];   // 2 KB, fragment order

    const int tid  = threadIdx.x;
    const int lane = tid & 31;
    const int wave = tid >> 5;     // 0..7
    const int wr   = wave & 3;     // wave row 0..3  (16 rows each)
    const int wc   = wave >> 2;    // wave col 0..1  (16 cols each)

    const int rowBase = blockIdx.y * TILE_M;
    const int colBase = blockIdx.x * TILE_N;

    const int hl = lane >> 4;      // lane half (0/1)
    const int lr = lane & 15;

    v8f acc = {};

    for (int k0 = 0; k0 < K; k0 += TILE_K) {
        // ---- stage A tile (64x32) as float2 pairs, 4 pairs/thread, clamped rows
        #pragma unroll
        for (int i = 0; i < 4; ++i) {
            int e  = tid + i * 256;        // pair index 0..1023
            int r  = e >> 4;               // row in tile 0..63
            int c  = (e & 15) << 1;        // even col in tile 0..30
            int gr = iclamp(rowBase + r, M - 1);
            int gk = k0 + c;
            float2 v2;
            if (GATHER) {
                if (gk < splitK) v2 = *(const float2*)(A  + (long)idx0[gr] * lda + gk);
                else             v2 = *(const float2*)(A1 + (long)idx1[gr] * lda + (gk - splitK));
            } else {
                v2 = *(const float2*)(A + (long)gr * lda + gk);
            }
            int fl = (r & 15) + 16 * ((c >> 3) & 1);
            int fe = (c & 7) + ((c & 16) ? 8 : 0);           // even
            union { h16 h[2]; unsigned u; } p;
            p.h[0] = (h16)v2.x; p.h[1] = (h16)v2.y;
            *(unsigned*)&As[r >> 4][fl][fe] = p.u;
        }
        // ---- stage B tile (32x32), 4 elems/thread, clamped cols
        #pragma unroll
        for (int i = 0; i < 4; ++i) {
            int e  = tid + i * 256;
            int r  = e >> 5;                         // k within tile, 0..31
            int c  = e & 31;                         // n within tile, 0..31
            int gc = iclamp(colBase + c, N - 1);
            float v = B[(long)(k0 + r) * ldb + gc];
            int fl = (c & 15) + 16 * ((r >> 3) & 1);
            int fe = (r & 7) + ((r & 16) ? 8 : 0);
            Bs[c >> 4][fl][fe] = (h16)v;
        }
        __syncthreads();

        // speculative prefetch of next A K-tile (global_prefetch_b8)
        if (!GATHER && (k0 + TILE_K) < K) {
            int pr = iclamp(rowBase + (tid >> 3), M - 1);
            __builtin_prefetch(&A[(long)pr * lda + (k0 + TILE_K)], 0, 0);
        }

        // ---- wide fragment loads: 32 contiguous bytes per lane
        v16h a_frag = *(const v16h*)(&As[wr][lane][0]);
        v16h b_frag = *(const v16h*)(&Bs[wc][lane][0]);

        acc = __builtin_amdgcn_wmma_f32_16x16x32_f16(
                false, a_frag, false, b_frag, (short)0, acc, false, false);
        __syncthreads();
    }

    // ---- epilogue: f32 C/D layout -> VGPR j is row (wr*16 + j + 8*hl), col = lane%16
    const int cCol = colBase + wc * 16 + lr;
    if (cCol < N) {
        float bval = bias ? bias[cCol] : 0.0f;
        float g = 1.0f, og = 0.0f;
        if (epilogue == EP_SKIP) {
            float s = 1.0f / (1.0f + expf(-gate[0]));
            g = s; og = 1.0f - s;
        }
        #pragma unroll
        for (int j = 0; j < 8; ++j) {
            int r = rowBase + wr * 16 + j + 8 * hl;
            if (r < M) {
                float v = acc[j] + bval;
                if (epilogue == EP_RELU)      v = fmaxf(v, 0.0f);
                else if (epilogue == EP_SKIP) v = g * v + og * oldx[(long)r * ldold + cCol];
                C[(long)r * ldc + cCol] = v;
            }
        }
    }
}

// ---------------------------------------------------------------------------
// BatchNorm (training mode, biased variance): per-column stats -> scale/shift
// ---------------------------------------------------------------------------
__global__ __launch_bounds__(256)
void bn_stats_kernel(const float* __restrict__ X, int ldx, int rows,
                     const float* __restrict__ g, const float* __restrict__ be,
                     float* __restrict__ scale, float* __restrict__ shift)
{
    const int col = blockIdx.x;
    float s = 0.0f, s2 = 0.0f;
    for (int r = threadIdx.x; r < rows; r += blockDim.x) {
        float v = X[(long)r * ldx + col];
        s += v; s2 += v * v;
    }
    __shared__ float sh[256], sh2[256];
    sh[threadIdx.x] = s; sh2[threadIdx.x] = s2;
    __syncthreads();
    for (int off = 128; off > 0; off >>= 1) {
        if ((int)threadIdx.x < off) {
            sh[threadIdx.x]  += sh[threadIdx.x + off];
            sh2[threadIdx.x] += sh2[threadIdx.x + off];
        }
        __syncthreads();
    }
    if (threadIdx.x == 0) {
        float inv = 1.0f / (float)rows;
        float mu  = sh[0] * inv;
        float var = sh2[0] * inv - mu * mu;
        float sc  = g[col] * rsqrtf(var + 1e-5f);
        scale[col] = sc;
        shift[col] = be[col] - mu * sc;
    }
}

__global__ __launch_bounds__(256)
void bn_apply_kernel(float* __restrict__ X, long n,
                     const float* __restrict__ scale, const float* __restrict__ shift,
                     int do_relu)
{
    long t = (long)blockIdx.x * blockDim.x + threadIdx.x;
    if (t >= n) return;
    int c = (int)(t & (HID - 1));
    float v = X[t] * scale[c] + shift[c];
    if (do_relu) v = fmaxf(v, 0.0f);
    X[t] = v;
}

// ---------------------------------------------------------------------------
// Edge attention kernels (per relation)
// ---------------------------------------------------------------------------
__device__ __forceinline__ unsigned f2ord(float f) {
    int i = __float_as_int(f);
    return (i < 0) ? (unsigned)(~i) : ((unsigned)i | 0x80000000u);
}
__device__ __forceinline__ float ord2f(unsigned u) {
    int i = (u & 0x80000000u) ? (int)(u & 0x7fffffffu) : ~(int)u;
    return __int_as_float(i);
}
#define ORD_NEG_INF 0x007FFFFFu   // f2ord(-inf)

__global__ __launch_bounds__(256)
void fill_u32_kernel(unsigned* p, unsigned v, long n) {
    long t = (long)blockIdx.x * blockDim.x + threadIdx.x;
    if (t < n) p[t] = v;
}
__global__ __launch_bounds__(256)
void fill_f32_kernel(float* p, float v, long n) {
    long t = (long)blockIdx.x * blockDim.x + threadIdx.x;
    if (t < n) p[t] = v;
}

// alpha[e,h] = <q_dst[dst[e],h,:], k_rel[src[e],h,:]> * p_rel[h] / sqrt(D)
// float4 vector loads: both row slices are 16B-aligned (offsets multiple of 32 floats)
__global__ __launch_bounds__(256)
void edge_alpha_kernel(const int* __restrict__ src, const int* __restrict__ dst,
                       const float* __restrict__ kr,        // [Nsrc, 64]
                       const float* __restrict__ kqv_dst,   // [Ndst, 192], q at +64
                       const float* __restrict__ p_rel,
                       float* __restrict__ alpha, int E)
{
    long t = (long)blockIdx.x * blockDim.x + threadIdx.x;
    if (t >= (long)E * NHEAD) return;
    int e = (int)(t >> 1), h = (int)(t & 1);
    const float4* kp = (const float4*)(kr      + (long)src[e] * HID + h * DHEAD);
    const float4* qp = (const float4*)(kqv_dst + (long)dst[e] * (3 * HID) + HID + h * DHEAD);
    float a = 0.0f;
    #pragma unroll
    for (int i = 0; i < DHEAD / 4; ++i) {
        float4 k4 = kp[i], q4 = qp[i];
        a += k4.x * q4.x + k4.y * q4.y + k4.z * q4.z + k4.w * q4.w;
    }
    alpha[t] = a * p_rel[h] * 0.17677669529663687f;   // 1/sqrt(32)
}

__global__ __launch_bounds__(256)
void seg_max_kernel(const float* __restrict__ alpha, const int* __restrict__ dst,
                    unsigned* __restrict__ mx, int E)
{
    long t = (long)blockIdx.x * blockDim.x + threadIdx.x;
    if (t >= (long)E * NHEAD) return;
    int e = (int)(t >> 1), h = (int)(t & 1);
    atomicMax(&mx[(long)dst[e] * NHEAD + h], f2ord(alpha[t]));
}

__global__ __launch_bounds__(256)
void mx_finalize_kernel(unsigned* __restrict__ mx, long n)
{
    long t = (long)blockIdx.x * blockDim.x + threadIdx.x;
    if (t >= n) return;
    float f = ord2f(mx[t]);
    if (!(fabsf(f) < 3.0e38f)) f = 0.0f;   // non-finite -> 0 (like reference)
    ((float*)mx)[t] = f;
}

// ex = exp(alpha - mx[dst]); alpha <- ex (in place); den[dst] += ex
__global__ __launch_bounds__(256)
void exp_den_kernel(float* __restrict__ alpha, const int* __restrict__ dst,
                    const float* __restrict__ mxf, float* __restrict__ den, int E)
{
    long t = (long)blockIdx.x * blockDim.x + threadIdx.x;
    if (t >= (long)E * NHEAD) return;
    int e = (int)(t >> 1), h = (int)(t & 1);
    long dh = (long)dst[e] * NHEAD + h;
    float ex = expf(alpha[t] - mxf[dh]);
    alpha[t] = ex;
    atomicAdd(&den[dh], ex);
}

// aggr[dst, c] += v_rel[src, c] * ex[e,h]/(den[dst,h]+1e-16)
__global__ __launch_bounds__(256)
void aggregate_kernel(const float* __restrict__ ex, const int* __restrict__ src,
                      const int* __restrict__ dst, const float* __restrict__ vr,
                      const float* __restrict__ den, float* __restrict__ aggr, int E)
{
    long t = (long)blockIdx.x * blockDim.x + threadIdx.x;
    if (t >= (long)E * HID) return;
    int e = (int)(t >> 6);
    int c = (int)(t & (HID - 1));
    int h = c >> 5;
    int d = dst[e], s = src[e];
    float w = ex[(long)e * NHEAD + h] / (den[(long)d * NHEAD + h] + 1e-16f);
    atomicAdd(&aggr[(long)d * HID + c], vr[(long)s * HID + c] * w);
}

// exact GELU (approximate=False): 0.5*x*(1+erf(x/sqrt(2)))
__global__ __launch_bounds__(256)
void gelu_kernel(float* __restrict__ x, long n)
{
    long t = (long)blockIdx.x * blockDim.x + threadIdx.x;
    if (t >= n) return;
    float v = x[t];
    x[t] = 0.5f * v * (1.0f + erff(v * 0.70710678118654752f));
}

// final MLP projection: logits[r] = h1[r,:] @ W2[:,0] + b2
__global__ __launch_bounds__(256)
void mlp_out_kernel(const float* __restrict__ h1, const float* __restrict__ W2,
                    const float* __restrict__ b2, float* __restrict__ logits, int L)
{
    long r = (long)blockIdx.x * blockDim.x + threadIdx.x;
    if (r >= L) return;
    float a = b2[0];
    const float* row = h1 + r * HID;
    #pragma unroll
    for (int j = 0; j < HID; ++j) a += row[j] * W2[j];
    logits[r] = a;
}

// ---------------------------------------------------------------------------
// Host orchestration
// ---------------------------------------------------------------------------
static inline int cdiv(long a, long b) { return (int)((a + b - 1) / b); }

static void run_gemm(hipStream_t st, const float* A, int lda,
                     const float* B, int ldb, const float* bias,
                     float* C, int ldc, int M, int N, int K,
                     int ep, const float* oldx, int ldold, const float* gate)
{
    dim3 grid(cdiv(N, TILE_N), cdiv(M, TILE_M));
    gemm_wmma_kernel<false><<<grid, 256, 0, st>>>(A, lda, nullptr, nullptr, nullptr, 0,
                                                  B, ldb, bias, C, ldc, M, N, K,
                                                  ep, oldx, ldold, gate);
}

static void run_gemm_gather(hipStream_t st, const float* A, const float* A1, int lda,
                            const int* i0, const int* i1, int splitK,
                            const float* B, int ldb, const float* bias,
                            float* C, int ldc, int M, int N, int K, int ep)
{
    dim3 grid(cdiv(N, TILE_N), cdiv(M, TILE_M));
    gemm_wmma_kernel<true><<<grid, 256, 0, st>>>(A, lda, A1, i0, i1, splitK,
                                                 B, ldb, bias, C, ldc, M, N, K,
                                                 ep, nullptr, 0, nullptr);
}

extern "C" void kernel_launch(void* const* d_in, const int* in_sizes, int n_in,
                              void* d_out, int out_size, void* d_ws, size_t ws_size,
                              hipStream_t stream)
{
    (void)in_sizes; (void)n_in; (void)out_size; (void)ws_size;
    const int ND = 100000, NP = 100000, E = 1000000, L = 500000;
    const int FD = 128, FP = 512;

    // ---- inputs (setup_inputs() dict insertion order, flattened) ----
    const float* x_drug = (const float*)d_in[0];
    const float* x_prot = (const float*)d_in[1];
    const int*   ei_dp  = (const int*)d_in[2];   // [2,E]: row0 = drug(src), row1 = prot(dst)
    const int*   ei_pd  = (const int*)d_in[3];   // [2,E]: row0 = prot(src), row1 = drug(dst)
    const int*   eli    = (const int*)d_in[4];   // [2,L]
    const float* W_d = (const float*)d_in[5],  *b_d = (const float*)d_in[6];
    const float* g_d = (const float*)d_in[7],  *be_d = (const float*)d_in[8];
    const float* W_p = (const float*)d_in[9],  *b_p = (const float*)d_in[10];
    const float* g_p = (const float*)d_in[11], *be_p = (const float*)d_in[12];
    // conv layers: base 13 and 29, 16 tensors each
    // mlp: 45..50
    const float* W1 = (const float*)d_in[45], *b1 = (const float*)d_in[46];
    const float* gM = (const float*)d_in[47], *beM = (const float*)d_in[48];
    const float* W2 = (const float*)d_in[49], *b2 = (const float*)d_in[50];

    // ---- outputs: [logits(L) | xd(ND*64) | xp(NP*64)] ----
    float* out_logits = (float*)d_out;
    float* out_xd = out_logits + L;
    float* out_xp = out_xd + (long)ND * HID;

    // ---- workspace layout ----
    float* ws = (float*)d_ws;
    size_t off = 0;
    auto alloc = [&](size_t n) { float* p = ws + off; off += n; return p; };
    float* xdA   = alloc((size_t)ND * HID);
    float* xpA   = alloc((size_t)NP * HID);
    float* xdB   = alloc((size_t)ND * HID);
    float* xpB   = alloc((size_t)NP * HID);
    float* kqv_d = alloc((size_t)ND * 3 * HID);
    float* kqv_p = alloc((size_t)NP * 3 * HID);
    float* kr    = alloc((size_t)ND * HID);      // max(ND,NP)*HID
    float* vr    = alloc((size_t)ND * HID);
    float* alpha = alloc((size_t)E * NHEAD);
    unsigned* mx = (unsigned*)alloc((size_t)NP * NHEAD);
    float* den   = alloc((size_t)NP * NHEAD);
    float* aggr  = alloc((size_t)NP * HID);
    float* h1    = alloc((size_t)L * HID);
    float* scale = alloc(HID);
    float* shift = alloc(HID);

    // ================= encoders: relu(bn(x @ W + b)) =================
    run_gemm(stream, x_drug, FD, W_d, HID, b_d, xdA, HID, ND, HID, FD,
             EP_BIAS, nullptr, 0, nullptr);
    bn_stats_kernel<<<HID, 256, 0, stream>>>(xdA, HID, ND, g_d, be_d, scale, shift);
    bn_apply_kernel<<<cdiv((long)ND * HID, 256), 256, 0, stream>>>(xdA, (long)ND * HID, scale, shift, 1);

    run_gemm(stream, x_prot, FP, W_p, HID, b_p, xpA, HID, NP, HID, FP,
             EP_BIAS, nullptr, 0, nullptr);
    bn_stats_kernel<<<HID, 256, 0, stream>>>(xpA, HID, NP, g_p, be_p, scale, shift);
    bn_apply_kernel<<<cdiv((long)NP * HID, 256), 256, 0, stream>>>(xpA, (long)NP * HID, scale, shift, 1);

    // ================= HGT layers =================
    auto relation = [&](const float* kqv_src, int Nsrc, const float* kqv_dst, int Ndst,
                        const int* src, const int* dst,
                        const float* a_rel, const float* m_rel, const float* p_rel)
    {
        // per-head K/V relation transforms (K at col 0, V at col 128 of kqv)
        for (int h = 0; h < NHEAD; ++h) {
            run_gemm(stream, kqv_src + h * DHEAD, 3 * HID,
                     a_rel + h * DHEAD * DHEAD, DHEAD, nullptr,
                     kr + h * DHEAD, HID, Nsrc, DHEAD, DHEAD, EP_BIAS, nullptr, 0, nullptr);
            run_gemm(stream, kqv_src + 2 * HID + h * DHEAD, 3 * HID,
                     m_rel + h * DHEAD * DHEAD, DHEAD, nullptr,
                     vr + h * DHEAD, HID, Nsrc, DHEAD, DHEAD, EP_BIAS, nullptr, 0, nullptr);
        }
        long nH = (long)Ndst * NHEAD;
        fill_u32_kernel<<<cdiv(nH, 256), 256, 0, stream>>>(mx, ORD_NEG_INF, nH);
        fill_f32_kernel<<<cdiv(nH, 256), 256, 0, stream>>>(den, 0.0f, nH);
        fill_f32_kernel<<<cdiv((long)Ndst * HID, 256), 256, 0, stream>>>(aggr, 0.0f, (long)Ndst * HID);

        long eh = (long)E * NHEAD;
        edge_alpha_kernel<<<cdiv(eh, 256), 256, 0, stream>>>(src, dst, kr, kqv_dst, p_rel, alpha, E);
        seg_max_kernel<<<cdiv(eh, 256), 256, 0, stream>>>(alpha, dst, mx, E);
        mx_finalize_kernel<<<cdiv(nH, 256), 256, 0, stream>>>(mx, nH);
        exp_den_kernel<<<cdiv(eh, 256), 256, 0, stream>>>(alpha, dst, (const float*)mx, den, E);
        aggregate_kernel<<<cdiv((long)E * HID, 256), 256, 0, stream>>>(alpha, src, dst, vr, den, aggr, E);
        gelu_kernel<<<cdiv((long)Ndst * HID, 256), 256, 0, stream>>>(aggr, (long)Ndst * HID);
    };

    auto hgt_layer = [&](const float* xd_in, const float* xp_in,
                         float* xd_out, float* xp_out, int base)
    {
        const float* Wkqv_d = (const float*)d_in[base + 0];
        const float* bkqv_d = (const float*)d_in[base + 1];
        const float* Wout_d = (const float*)d_in[base + 2];
        const float* bout_d = (const float*)d_in[base + 3];
        const float* skip_d = (const float*)d_in[base + 4];
        const float* Wkqv_p = (const float*)d_in[base + 5];
        const float* bkqv_p = (const float*)d_in[base + 6];
        const float* Wout_p = (const float*)d_in[base + 7];
        const float* bout_p = (const float*)d_in[base + 8];
        const float* skip_p = (const float*)d_in[base + 9];
        const float* a_dp = (const float*)d_in[base + 10];
        const float* m_dp = (const float*)d_in[base + 11];
        const float* p_dp = (const float*)d_in[base + 12];
        const float* a_pd = (const float*)d_in[base + 13];
        const float* m_pd = (const float*)d_in[base + 14];
        const float* p_pd = (const float*)d_in[base + 15];

        run_gemm(stream, xd_in, HID, Wkqv_d, 3 * HID, bkqv_d,
                 kqv_d, 3 * HID, ND, 3 * HID, HID, EP_BIAS, nullptr, 0, nullptr);
        run_gemm(stream, xp_in, HID, Wkqv_p, 3 * HID, bkqv_p,
                 kqv_p, 3 * HID, NP, 3 * HID, HID, EP_BIAS, nullptr, 0, nullptr);

        // relation drug->protein: aggregates into protein nodes
        relation(kqv_d, ND, kqv_p, NP, ei_dp, ei_dp + E, a_dp, m_dp, p_dp);
        run_gemm(stream, aggr, HID, Wout_p, HID, bout_p,
                 xp_out, HID, NP, HID, HID, EP_SKIP, xp_in, HID, skip_p);

        // relation protein->drug: aggregates into drug nodes
        relation(kqv_p, NP, kqv_d, ND, ei_pd, ei_pd + E, a_pd, m_pd, p_pd);
        run_gemm(stream, aggr, HID, Wout_d, HID, bout_d,
                 xd_out, HID, ND, HID, HID, EP_SKIP, xd_in, HID, skip_d);
    };

    hgt_layer(xdA, xpA, xdB, xpB, 13);              // layer 0 -> scratch
    hgt_layer(xdB, xpB, out_xd, out_xp, 29);        // layer 1 -> d_out slices

    // ================= link MLP =================
    // h1 = relu(concat(xd[eli0], xp[eli1]) @ W1 + b1)   (gather fused into GEMM)
    run_gemm_gather(stream, out_xd, out_xp, HID, eli, eli + L, HID,
                    W1, HID, b1, h1, HID, L, HID, 2 * HID, EP_RELU);
    bn_stats_kernel<<<HID, 256, 0, stream>>>(h1, HID, L, gM, beM, scale, shift);
    bn_apply_kernel<<<cdiv((long)L * HID, 256), 256, 0, stream>>>(h1, (long)L * HID, scale, shift, 0);
    mlp_out_kernel<<<cdiv((long)L, 256), 256, 0, stream>>>(h1, W2, b2, out_logits, L);
}
